// SelfAttention_72060961292401
// MI455X (gfx1250) — matrix operand order repaired and checked by reading, not compiled, not run on hardware
//
#include <hip/hip_runtime.h>
#include <hip/hip_bf16.h>

// ---------------------------------------------------------------------------
// Problem constants (from the reference)
// ---------------------------------------------------------------------------
#define NB   8
#define TS   2048
#define CD   128
#define SCALE 0.08838834764831845f   // 1/sqrt(128)

typedef __attribute__((ext_vector_type(16))) __bf16 v16bf;
typedef __attribute__((ext_vector_type(8)))  __bf16 v8bf;
typedef __attribute__((ext_vector_type(8)))  float  v8f;
typedef __attribute__((ext_vector_type(4)))  float  v4f;

// round-to-nearest-even f32 -> bf16
static __device__ __forceinline__ __bf16 cvt_bf16(float f) {
    unsigned u = __float_as_uint(f);
    u += 0x7FFFu + ((u >> 16) & 1u);
    unsigned short h = (unsigned short)(u >> 16);
    return __builtin_bit_cast(__bf16, h);
}

static __device__ __forceinline__ v8f wmma_bf16(v16bf a, v16bf b, v8f c) {
    // D = A(16x32 bf16) * B(32x16 bf16) + C(16x16 f32)
    return __builtin_amdgcn_wmma_f32_16x16x32_bf16(
        false, a, false, b, (short)0, c, false, false);
}

// A-fragment (16x32, row-major source, `stride` bf16 elems per row), K-step base k0.
// ISA layout: lanes 0-15 (M=lane): elems 0..7 = K[k0..k0+7], 8..15 = K[k0+16..k0+23]
//             lanes 16-31 (M=lane-16): K[k0+8..15] and K[k0+24..31]
static __device__ __forceinline__ v16bf load_a_frag(const __bf16* rowbase, int stride,
                                                    int k0, int lane) {
    int half = lane >> 4, m = lane & 15;
    const __bf16* r = rowbase + m * stride;
    v8bf lo = *(const v8bf*)(r + k0 + 8 * half);
    v8bf hi = *(const v8bf*)(r + k0 + 16 + 8 * half);
    return __builtin_shufflevector(lo, hi, 0,1,2,3,4,5,6,7,8,9,10,11,12,13,14,15);
}

// B-fragment (32x16): B[k][n] = src[n][k], i.e. B column n = source row n (contiguous).
// lanes 0-15 (N=lane): K[k0..k0+7], K[k0+16..k0+23]; lanes 16-31 (N=lane-16): K[k0+8..15], K[k0+24..31]
static __device__ __forceinline__ v16bf load_b_frag(const __bf16* colsrc, int stride,
                                                    int k0, int lane) {
    int half = lane >> 4, m = lane & 15;
    const __bf16* r = colsrc + m * stride;
    v8bf lo = *(const v8bf*)(r + k0 + 8 * half);
    v8bf hi = *(const v8bf*)(r + k0 + 16 + 8 * half);
    return __builtin_shufflevector(lo, hi, 0,1,2,3,4,5,6,7,8,9,10,11,12,13,14,15);
}

// ---------------------------------------------------------------------------
// f32 -> bf16 conversion kernels
// ---------------------------------------------------------------------------
__global__ void cvt_kernel(const float* __restrict__ src, __bf16* __restrict__ dst, int n) {
    for (int i = blockIdx.x * blockDim.x + threadIdx.x; i < n; i += gridDim.x * blockDim.x)
        dst[i] = cvt_bf16(src[i]);
}

__global__ void zero4_kernel(float* __restrict__ p, long long n4) {
    v4f z = {};
    long long stride = (long long)gridDim.x * blockDim.x;
    for (long long i = blockIdx.x * (long long)blockDim.x + threadIdx.x; i < n4; i += stride)
        ((v4f*)p)[i] = z;
}

// ---------------------------------------------------------------------------
// Q/K projection:  Q = E @ Wq^T, K = E @ Wk^T  (bf16 in, f32 acc, bf16 out)
// grid = (NB*TS)/16 row tiles, block = 256 (8 waves -> 8 column tiles of 16)
// ---------------------------------------------------------------------------
__global__ void __launch_bounds__(256) proj_qk_kernel(const __bf16* __restrict__ Ebf,
                                                      const __bf16* __restrict__ Wqbf,
                                                      const __bf16* __restrict__ Wkbf,
                                                      __bf16* __restrict__ Qbf,
                                                      __bf16* __restrict__ Kbf) {
    int lane = threadIdx.x & 31;
    int n0   = (threadIdx.x >> 5) * 16;       // output-column tile
    int rt   = blockIdx.x;                    // 16-row tile over NB*TS rows
    const __bf16* Arow = Ebf + rt * 16 * CD;

    v16bf a[4];
    #pragma unroll
    for (int s = 0; s < 4; ++s) a[s] = load_a_frag(Arow, CD, 32 * s, lane);

    v8f accq = {}, acck = {};
    #pragma unroll
    for (int s = 0; s < 4; ++s) {
        v16bf bq = load_b_frag(Wqbf + n0 * CD, CD, 32 * s, lane);
        v16bf bk = load_b_frag(Wkbf + n0 * CD, CD, 32 * s, lane);
        accq = wmma_bf16(a[s], bq, accq);
        acck = wmma_bf16(a[s], bk, acck);
    }

    int half = lane >> 4, m = lane & 15;
    #pragma unroll
    for (int r = 0; r < 8; ++r) {
        int row = rt * 16 + r + 8 * half;
        Qbf[row * CD + n0 + m] = cvt_bf16(accq[r]);
        Kbf[row * CD + n0 + m] = cvt_bf16(acck[r]);
    }
}

// ---------------------------------------------------------------------------
// Scores + top-3 + softmax + attention scatter + context gather.
// One wave per (batch n, 16-row q tile): 1024 waves total.
// ---------------------------------------------------------------------------
__global__ void __launch_bounds__(256) scores_kernel(const __bf16* __restrict__ Qbf,
                                                     const __bf16* __restrict__ Kbf,
                                                     const float*  __restrict__ Ef32,
                                                     __bf16* __restrict__ out1bf,
                                                     float* __restrict__ att) {
    int lane = threadIdx.x & 31;
    int wid  = blockIdx.x * 8 + (threadIdx.x >> 5);   // 0..1023
    int n    = wid >> 7;                              // batch
    int q0   = (wid & 127) * 16;                      // q-row tile within batch
    int half = lane >> 4, m = lane & 15;

    const __bf16* Arow = Qbf + (n * TS + q0) * CD;
    v16bf a[4];
    #pragma unroll
    for (int s = 0; s < 4; ++s) a[s] = load_a_frag(Arow, CD, 32 * s, lane);

    // per-lane running top-3 for the 8 rows this lane carries in the D fragment
    float tv0[8], tv1[8], tv2[8];
    int   ti0[8], ti1[8], ti2[8];
    #pragma unroll
    for (int r = 0; r < 8; ++r) {
        tv0[r] = tv1[r] = tv2[r] = -__builtin_inff();
        ti0[r] = ti1[r] = ti2[r] = 0;
    }

    const __bf16* Kbase = Kbf + n * TS * CD;

    for (int t0 = 0; t0 < TS; t0 += 16) {
        if (t0 + 16 < TS)   // pull next B tile toward the caches
            __builtin_prefetch(Kbase + (t0 + 16 + m) * CD, 0, 0);

        v8f acc = {};
        #pragma unroll
        for (int s = 0; s < 4; ++s) {
            v16bf b = load_b_frag(Kbase + t0 * CD, CD, 32 * s, lane);
            acc = wmma_bf16(a[s], b, acc);
        }

        int col = t0 + m;   // score column this lane holds
        #pragma unroll
        for (int r = 0; r < 8; ++r) {
            float v = acc[r];
            bool g0 = v > tv0[r];
            bool g1 = v > tv1[r];
            bool g2 = v > tv2[r];
            float nv2 = g1 ? tv1[r] : (g2 ? v : tv2[r]);
            int   ni2 = g1 ? ti1[r] : (g2 ? col : ti2[r]);
            float nv1 = g0 ? tv0[r] : (g1 ? v : tv1[r]);
            int   ni1 = g0 ? ti0[r] : (g1 ? col : ti1[r]);
            tv0[r] = g0 ? v   : tv0[r];
            ti0[r] = g0 ? col : ti0[r];
            tv1[r] = nv1; ti1[r] = ni1;
            tv2[r] = nv2; ti2[r] = ni2;
        }
    }

    // merge top-3 lists across the 16 lanes of each half (xor butterflies stay in-half)
    #pragma unroll
    for (int mask = 1; mask < 16; mask <<= 1) {
        #pragma unroll
        for (int r = 0; r < 8; ++r) {
            float ov[3]; int oi[3];
            ov[0] = __shfl_xor(tv0[r], mask, 32); oi[0] = __shfl_xor(ti0[r], mask, 32);
            ov[1] = __shfl_xor(tv1[r], mask, 32); oi[1] = __shfl_xor(ti1[r], mask, 32);
            ov[2] = __shfl_xor(tv2[r], mask, 32); oi[2] = __shfl_xor(ti2[r], mask, 32);
            #pragma unroll
            for (int j = 0; j < 3; ++j) {
                float v = ov[j]; int c = oi[j];
                bool g0 = v > tv0[r];
                bool g1 = v > tv1[r];
                bool g2 = v > tv2[r];
                float nv2 = g1 ? tv1[r] : (g2 ? v : tv2[r]);
                int   ni2 = g1 ? ti1[r] : (g2 ? c : ti2[r]);
                float nv1 = g0 ? tv0[r] : (g1 ? v : tv1[r]);
                int   ni1 = g0 ? ti0[r] : (g1 ? c : ti1[r]);
                tv0[r] = g0 ? v : tv0[r];
                ti0[r] = g0 ? c : ti0[r];
                tv1[r] = nv1; ti1[r] = ni1;
                tv2[r] = nv2; ti2[r] = ni2;
            }
        }
    }

    // softmax over the 3 survivors (scale applied here; top-3 is scale-invariant)
    float p0[8], p1[8], p2[8];
    #pragma unroll
    for (int r = 0; r < 8; ++r) {
        float e0 = 1.0f;
        float e1 = __expf((tv1[r] - tv0[r]) * SCALE);
        float e2 = __expf((tv2[r] - tv0[r]) * SCALE);
        float inv = 1.0f / (e0 + e1 + e2);
        p0[r] = e0 * inv; p1[r] = e1 * inv; p2[r] = e2 * inv;
    }

    // scatter the 3 probabilities per row into the (pre-zeroed) dense attention
    float* abase = att + (long long)n * TS * TS;
    #pragma unroll
    for (int r = 0; r < 8; ++r) {
        if (m == r) {
            float* arow = abase + (long long)(q0 + r + 8 * half) * TS;
            arow[ti0[r]] = p0[r];
            arow[ti1[r]] = p1[r];
            arow[ti2[r]] = p2[r];
        }
    }

    // context = sum_j p_j * V[idx_j]  (V = raw f32 embeddings); store bf16 for out-proj
    const float* Vbase = Ef32 + (long long)n * TS * CD;
    #pragma unroll
    for (int r = 0; r < 8; ++r) {
        int row = q0 + r + 8 * half;
        const float* V0 = Vbase + (long long)ti0[r] * CD;
        const float* V1 = Vbase + (long long)ti1[r] * CD;
        const float* V2 = Vbase + (long long)ti2[r] * CD;
        int c = m * 8;
        v8bf ov;
        #pragma unroll
        for (int j = 0; j < 8; ++j) {
            float val = p0[r] * V0[c + j] + p1[r] * V1[c + j] + p2[r] * V2[c + j];
            ov[j] = cvt_bf16(val);
        }
        *(v8bf*)(out1bf + ((long long)(n * TS + row)) * CD + c) = ov;
    }
}

// ---------------------------------------------------------------------------
// Output projection:  out = ctx @ Wo^T + bo   (f32 result)
// ---------------------------------------------------------------------------
__global__ void __launch_bounds__(256) out_proj_kernel(const __bf16* __restrict__ Cbf,
                                                       const __bf16* __restrict__ Wobf,
                                                       const float* __restrict__ bo,
                                                       float* __restrict__ out) {
    int lane = threadIdx.x & 31;
    int n0   = (threadIdx.x >> 5) * 16;
    int rt   = blockIdx.x;
    const __bf16* Arow = Cbf + rt * 16 * CD;

    v16bf a[4];
    #pragma unroll
    for (int s = 0; s < 4; ++s) a[s] = load_a_frag(Arow, CD, 32 * s, lane);

    v8f acc = {};
    #pragma unroll
    for (int s = 0; s < 4; ++s) {
        v16bf b = load_b_frag(Wobf + n0 * CD, CD, 32 * s, lane);
        acc = wmma_bf16(a[s], b, acc);
    }

    int half = lane >> 4, m = lane & 15;
    float bias = bo[n0 + m];
    #pragma unroll
    for (int r = 0; r < 8; ++r) {
        int row = rt * 16 + r + 8 * half;
        out[(long long)row * CD + n0 + m] = acc[r] + bias;
    }
}

// ---------------------------------------------------------------------------
// Launcher
// ---------------------------------------------------------------------------
extern "C" void kernel_launch(void* const* d_in, const int* in_sizes, int n_in,
                              void* d_out, int out_size, void* d_ws, size_t ws_size,
                              hipStream_t stream) {
    const float* E  = (const float*)d_in[0];   // (8,2048,128)
    // d_in[1] = ac (unused by forward)
    const float* Wq = (const float*)d_in[2];   // (128,128)
    const float* Wk = (const float*)d_in[3];
    const float* Wo = (const float*)d_in[4];
    const float* bo = (const float*)d_in[5];   // (128,)

    float* out = (float*)d_out;                        // (8,2048,128) f32
    float* att = (float*)d_out + (long long)NB * TS * CD;  // (8,1,2048,2048) f32

    const int NE = NB * TS * CD;   // 2,097,152
    const int NW = CD * CD;        // 16,384

    // workspace carve-up (bytes)
    char* ws = (char*)d_ws;
    __bf16* Ebf   = (__bf16*)(ws);                       //  4 MB
    __bf16* Qbf   = (__bf16*)(ws + (size_t)NE * 2);      //  4 MB
    __bf16* Kbf   = (__bf16*)(ws + (size_t)NE * 4);      //  4 MB
    __bf16* Cbf   = (__bf16*)(ws + (size_t)NE * 6);      //  4 MB (context)
    __bf16* Wqbf  = (__bf16*)(ws + (size_t)NE * 8);
    __bf16* Wkbf  = (__bf16*)(ws + (size_t)NE * 8 + (size_t)NW * 2);
    __bf16* Wobf  = (__bf16*)(ws + (size_t)NE * 8 + (size_t)NW * 4);

    // 1) f32 -> bf16 conversions
    cvt_kernel<<<2048, 256, 0, stream>>>(E,  Ebf,  NE);
    cvt_kernel<<<64,   256, 0, stream>>>(Wq, Wqbf, NW);
    cvt_kernel<<<64,   256, 0, stream>>>(Wk, Wkbf, NW);
    cvt_kernel<<<64,   256, 0, stream>>>(Wo, Wobf, NW);

    // 2) zero the dense attention output (33,554,432 f32 = 8,388,608 float4)
    zero4_kernel<<<8192, 256, 0, stream>>>(att, (long long)NB * TS * TS / 4);

    // 3) Q/K projections (WMMA bf16 -> f32)
    proj_qk_kernel<<<(NB * TS) / 16, 256, 0, stream>>>(Ebf, Wqbf, Wkbf, Qbf, Kbf);

    // 4) scores + top-3 + softmax + scatter + context gather
    scores_kernel<<<(NB * TS / 16) / 8, 256, 0, stream>>>(Qbf, Kbf, E, Cbf, att);

    // 5) output projection + bias (WMMA bf16 -> f32)
    out_proj_kernel<<<(NB * TS) / 16, 256, 0, stream>>>(Cbf, Wobf, bo, out);
}
